// CustomStellarModel2_48722109006552
// MI455X (gfx1250) — matrix operand order repaired and checked by reading, not compiled, not run on hardware
//
#include <hip/hip_runtime.h>
#include <hip/hip_bf16.h>

#define NN 100000
#define EE 1600000
#define IN_F 64
#define HH 128
#define CCLS 20

typedef __attribute__((ext_vector_type(16))) __bf16 v16bf;
typedef __attribute__((ext_vector_type(8)))  float  v8f;
typedef unsigned int u32x4 __attribute__((ext_vector_type(4)));
typedef float f32x4 __attribute__((ext_vector_type(4)));

__device__ __forceinline__ unsigned short f2bf(float f) {
    unsigned int u = __float_as_uint(f);
    u += 0x7FFFu + ((u >> 16) & 1u);   // round-to-nearest-even
    return (unsigned short)(u >> 16);
}

// ---------------- elementwise f32 -> bf16 conversion ----------------
__global__ void cvt_f32_to_bf16(const float* __restrict__ s, unsigned short* __restrict__ d, int n) {
    int i = blockIdx.x * 256 + threadIdx.x;
    if (i < n) d[i] = f2bf(s[i]);
}

// ---------------- WMMA GEMM: Y[M,ldY] = X[M,K] @ W[Fout,K]^T + bias ----------------
// 4 waves per 128-thread block; each wave computes one 16x16 tile via v_wmma_f32_16x16x32_bf16.
// Fragment layouts per CDNA5 ISA 7.12.2:
//   A 16x32 bf16: lane L (row m0+(L&15)): VGPR0-3 = K {off..off+7}, VGPR4-7 = K {16+off..23+off}, off=(L>>4)*8
//   B 32x16 bf16: lane L (col n0+(L&15)): same K order as A: VGPR0-3 = K {off..off+7}, VGPR4-7 = K {16+off..23+off}
//   D 16x16 f32 : lane L col n0+(L&15); VGPR v -> row m0 + v + (L>>4)*8
template<bool RELU, bool OUTBF16>
__global__ void wmma_gemm(const unsigned short* __restrict__ X,
                          const unsigned short* __restrict__ W,
                          const float* __restrict__ bias,
                          float* __restrict__ Yf,
                          unsigned short* __restrict__ Yb,
                          int K, int ldY)
{
    const int lane = threadIdx.x & 31;
    const int wave = threadIdx.x >> 5;
    const int m0   = blockIdx.x * 16;
    const int n0   = (blockIdx.y * 4 + wave) * 16;
    const int half = lane >> 4;
    const int l15  = lane & 15;

    const unsigned short* aPtr = X + (size_t)(m0 + l15) * K + half * 8;
    const unsigned short* bPtr = W + (size_t)(n0 + l15) * K + half * 8;

    v8f acc = {0.f, 0.f, 0.f, 0.f, 0.f, 0.f, 0.f, 0.f};
    for (int kc = 0; kc < K; kc += 32) {
        union { v16bf v; u32x4 q[2]; } A, B;
        A.q[0] = *(const u32x4*)(aPtr + kc);        // K = kc+off   .. +7
        A.q[1] = *(const u32x4*)(aPtr + kc + 16);   // K = kc+16+off.. +7
        const u32x4* pb = (const u32x4*)(bPtr + kc);
        B.q[0] = pb[0];                             // K = kc+off   .. +7
        B.q[1] = pb[2];                             // K = kc+16+off.. +7
        acc = __builtin_amdgcn_wmma_f32_16x16x32_bf16(
                  false, A.v, false, B.v, (short)0, acc, false, false);
    }

    const int nCol = n0 + l15;
    const float bb = bias ? bias[nCol] : 0.0f;
#pragma unroll
    for (int v = 0; v < 8; ++v) {
        int row = m0 + half * 8 + v;
        float val = acc[v] + bb;
        if (RELU) val = fmaxf(val, 0.0f);
        if (OUTBF16) Yb[(size_t)row * ldY + nCol] = f2bf(val);
        else         Yf[(size_t)row * ldY + nCol] = val;
    }
}

// ---------------- out = relu(gamma_s * xls + beta_s) ----------------
__global__ void make_out(const float* __restrict__ bgs, const float* __restrict__ xls,
                         float* __restrict__ outb)
{
    int i = blockIdx.x * 256 + threadIdx.x;
    if (i >= NN * HH) return;
    int row = i >> 7, col = i & 127;
    float beta  = bgs[(size_t)row * 256 + col];
    float gamma = bgs[(size_t)row * 256 + 128 + col];
    outb[i] = fmaxf(gamma * xls[i] + beta, 0.0f);
}

// ---------------- edge pass 1: msg = relu(gamma[dst]*xl[src]+beta[dst]); scatter-add ----------------
__global__ void edge_msg(const long long* __restrict__ ei,
                         const float* __restrict__ bg,   // [N,256] beta|gamma
                         const float* __restrict__ xl,   // [N,128]
                         float* __restrict__ s,          // [N,128] accum
                         float* __restrict__ cnt)        // [N]
{
    int t = blockIdx.x * 256 + threadIdx.x;
    if (t >= EE * 32) return;
    int e = t >> 5, c = t & 31;
    int src = (int)ei[e];
    int dst = (int)ei[EE + e];
    f32x4 be = *(const f32x4*)(bg + (size_t)dst * 256 + c * 4);
    f32x4 ga = *(const f32x4*)(bg + (size_t)dst * 256 + 128 + c * 4);
    f32x4 xv = *(const f32x4*)(xl + (size_t)src * 128 + c * 4);
    f32x4 m = ga * xv + be;
    float* sp = s + (size_t)dst * 128 + c * 4;
    unsafeAtomicAdd(sp + 0, fmaxf(m.x, 0.0f));
    unsafeAtomicAdd(sp + 1, fmaxf(m.y, 0.0f));
    unsafeAtomicAdd(sp + 2, fmaxf(m.z, 0.0f));
    unsafeAtomicAdd(sp + 3, fmaxf(m.w, 0.0f));
    if (c == 0) unsafeAtomicAdd(cnt + dst, 1.0f);
}

// ---------------- x2 = relu(out + s/max(cnt,1)) -> bf16 ----------------
__global__ void make_x2(const float* __restrict__ outb, const float* __restrict__ s,
                        const float* __restrict__ cnt, unsigned short* __restrict__ x2b)
{
    int i = blockIdx.x * 256 + threadIdx.x;
    if (i >= NN * HH) return;
    int row = i >> 7;
    float agg = s[i] / fmaxf(cnt[row], 1.0f);
    x2b[i] = f2bf(fmaxf(outb[i] + agg, 0.0f));
}

// ---------------- edge pass 2: eta = sigmoid(k[dst]+q[src]); s2 += eta*v[src] ----------------
__global__ void edge_att(const long long* __restrict__ ei,
                         const float* __restrict__ kqvs,   // [N,512]: k|q|v|skip
                         float* __restrict__ s2)
{
    int t = blockIdx.x * 256 + threadIdx.x;
    if (t >= EE * 32) return;
    int e = t >> 5, c = t & 31;
    int src = (int)ei[e];
    int dst = (int)ei[EE + e];
    f32x4 kk = *(const f32x4*)(kqvs + (size_t)dst * 512 + c * 4);
    f32x4 qq = *(const f32x4*)(kqvs + (size_t)src * 512 + 128 + c * 4);
    f32x4 vv = *(const f32x4*)(kqvs + (size_t)src * 512 + 256 + c * 4);
    f32x4 z = kk + qq;
    f32x4 m;
    m.x = vv.x / (1.0f + __expf(-z.x));
    m.y = vv.y / (1.0f + __expf(-z.y));
    m.z = vv.z / (1.0f + __expf(-z.z));
    m.w = vv.w / (1.0f + __expf(-z.w));
    float* sp = s2 + (size_t)dst * 128 + c * 4;
    unsafeAtomicAdd(sp + 0, m.x);
    unsafeAtomicAdd(sp + 1, m.y);
    unsafeAtomicAdd(sp + 2, m.z);
    unsafeAtomicAdd(sp + 3, m.w);
}

// ---------------- x3 = relu(s2 + skip_pre); logits = x3 @ Wfc^T + bfc (duplicated) ----------------
__global__ void final_fc(const float* __restrict__ s2, const float* __restrict__ kqvs,
                         const float* __restrict__ Wfc, const float* __restrict__ bfc,
                         float* __restrict__ out)
{
    __shared__ float w[CCLS * HH];
    __shared__ float b[CCLS];
    __shared__ float x3[8 * HH];
    const int tid = threadIdx.x;   // 160 threads = 8 nodes x 20 classes
    for (int i = tid; i < CCLS * HH; i += 160) w[i] = Wfc[i];
    if (tid < CCLS) b[tid] = bfc[tid];
    const int nodeBase = blockIdx.x * 8;
    for (int i = tid; i < 8 * HH; i += 160) {
        int nl = i >> 7, col = i & 127;
        int row = nodeBase + nl;
        float v = s2[(size_t)row * 128 + col] + kqvs[(size_t)row * 512 + 384 + col];
        x3[i] = fmaxf(v, 0.0f);
    }
    __syncthreads();
    const int nl = tid / CCLS, cc = tid % CCLS;
    float sum = b[cc];
#pragma unroll 8
    for (int j = 0; j < HH; ++j) sum += w[cc * HH + j] * x3[nl * HH + j];
    const int row = nodeBase + nl;
    out[(size_t)row * CCLS + cc] = sum;
    out[(size_t)NN * CCLS + (size_t)row * CCLS + cc] = sum;
}

extern "C" void kernel_launch(void* const* d_in, const int* in_sizes, int n_in,
                              void* d_out, int out_size, void* d_ws, size_t ws_size,
                              hipStream_t stream)
{
    const float*     x    = (const float*)d_in[0];
    const long long* ei   = (const long long*)d_in[1];
    const float*     W1   = (const float*)d_in[2];
    const float*     b1   = (const float*)d_in[3];
    const float*     Wl   = (const float*)d_in[4];
    const float*     Wf   = (const float*)d_in[5];
    const float*     bf   = (const float*)d_in[6];
    const float*     Wls  = (const float*)d_in[7];
    const float*     Wfs  = (const float*)d_in[8];
    const float*     bfs  = (const float*)d_in[9];
    const float*     Wk   = (const float*)d_in[10];
    const float*     bk   = (const float*)d_in[11];
    const float*     Wq   = (const float*)d_in[12];
    const float*     bq   = (const float*)d_in[13];
    const float*     Wv   = (const float*)d_in[14];
    const float*     bv   = (const float*)d_in[15];
    const float*     Wsk  = (const float*)d_in[16];
    const float*     bres = (const float*)d_in[17];
    const float*     Wfc  = (const float*)d_in[18];
    const float*     bfc  = (const float*)d_in[19];

    char* ws = (char*)d_ws;
    // workspace layout (bytes), all 16B aligned
    unsigned short* xb   = (unsigned short*)(ws + 0);            // N*64  bf16  (12.8 MB)
    unsigned short* x1b  = (unsigned short*)(ws + 12800000);     // N*128 bf16  (25.6 MB)
    float* bufA          = (float*)(ws + 38400000);              // N*256 f32   (102.4 MB) bgs/bg, kqvs[0:256]
    float* bufB          = (float*)(ws + 140800000);             // N*128 f32   (51.2 MB)  xls/xl, kqvs[256:384]
    float* bufC          = (float*)(ws + 192000000);             // N*128 f32   (51.2 MB)  out,    kqvs[384:512]
    float* kqvs          = bufA;                                 // [N,512] spans A+B+C
    float* bufS          = (float*)(ws + 243200000);             // N*128 f32   (51.2 MB)  s / s2
    float* cnt           = (float*)(ws + 294400000);             // N f32
    unsigned short* x2b  = (unsigned short*)(ws + 294800000);    // N*128 bf16
    unsigned short* wp   = (unsigned short*)(ws + 320400000);    // bf16 weight pool
    unsigned short* W1b  = wp;            // 128*64
    unsigned short* Wfsb = wp + 8192;     // 256*128
    unsigned short* Wlsb = wp + 40960;    // 128*128
    unsigned short* Wfb  = wp + 57344;    // 256*128
    unsigned short* Wlb  = wp + 90112;    // 128*128
    unsigned short* Wkb  = wp + 106496;   // 128*128
    unsigned short* Wqb  = wp + 122880;
    unsigned short* Wvb  = wp + 139264;
    unsigned short* Wskb = wp + 155648;

    auto nb = [](int n) { return (n + 255) / 256; };

    // ---- convert inputs/weights to bf16 ----
    cvt_f32_to_bf16<<<nb(NN * IN_F), 256, 0, stream>>>(x,   xb,   NN * IN_F);
    cvt_f32_to_bf16<<<nb(8192),  256, 0, stream>>>(W1,  W1b,  8192);
    cvt_f32_to_bf16<<<nb(32768), 256, 0, stream>>>(Wfs, Wfsb, 32768);
    cvt_f32_to_bf16<<<nb(16384), 256, 0, stream>>>(Wls, Wlsb, 16384);
    cvt_f32_to_bf16<<<nb(32768), 256, 0, stream>>>(Wf,  Wfb,  32768);
    cvt_f32_to_bf16<<<nb(16384), 256, 0, stream>>>(Wl,  Wlb,  16384);
    cvt_f32_to_bf16<<<nb(16384), 256, 0, stream>>>(Wk,  Wkb,  16384);
    cvt_f32_to_bf16<<<nb(16384), 256, 0, stream>>>(Wq,  Wqb,  16384);
    cvt_f32_to_bf16<<<nb(16384), 256, 0, stream>>>(Wv,  Wvb,  16384);
    cvt_f32_to_bf16<<<nb(16384), 256, 0, stream>>>(Wsk, Wskb, 16384);

    const int MT = NN / 16;  // 6250 row tiles

    // ---- x1 = relu(x @ W1^T + b1)  (bf16 out) ----
    wmma_gemm<true, true><<<dim3(MT, 2), 128, 0, stream>>>(xb, W1b, b1, nullptr, x1b, IN_F, HH);

    // ---- stage-1 linears ----
    wmma_gemm<false, false><<<dim3(MT, 4), 128, 0, stream>>>(x1b, Wfsb, bfs, bufA, nullptr, HH, 256); // bgs
    wmma_gemm<false, false><<<dim3(MT, 2), 128, 0, stream>>>(x1b, Wlsb, nullptr, bufB, nullptr, HH, 128); // xls
    make_out<<<nb(NN * HH), 256, 0, stream>>>(bufA, bufB, bufC);                                      // out
    wmma_gemm<false, false><<<dim3(MT, 4), 128, 0, stream>>>(x1b, Wfb, bf, bufA, nullptr, HH, 256);   // bg
    wmma_gemm<false, false><<<dim3(MT, 2), 128, 0, stream>>>(x1b, Wlb, nullptr, bufB, nullptr, HH, 128); // xl

    // ---- edge pass 1 (mean aggregation) ----
    hipMemsetAsync(bufS, 0, (size_t)NN * HH * sizeof(float), stream);
    hipMemsetAsync(cnt,  0, (size_t)NN * sizeof(float), stream);
    edge_msg<<<nb(EE * 32), 256, 0, stream>>>(ei, bufA, bufB, bufS, cnt);
    make_x2<<<nb(NN * HH), 256, 0, stream>>>(bufC, bufS, cnt, x2b);

    // ---- stage-2 linears: k|q|v|skip into [N,512] ----
    wmma_gemm<false, false><<<dim3(MT, 2), 128, 0, stream>>>(x2b, Wkb,  bk,   kqvs + 0,   nullptr, HH, 512);
    wmma_gemm<false, false><<<dim3(MT, 2), 128, 0, stream>>>(x2b, Wqb,  bq,   kqvs + 128, nullptr, HH, 512);
    wmma_gemm<false, false><<<dim3(MT, 2), 128, 0, stream>>>(x2b, Wvb,  bv,   kqvs + 256, nullptr, HH, 512);
    wmma_gemm<false, false><<<dim3(MT, 2), 128, 0, stream>>>(x2b, Wskb, bres, kqvs + 384, nullptr, HH, 512);

    // ---- edge pass 2 (gated aggregation) ----
    hipMemsetAsync(bufS, 0, (size_t)NN * HH * sizeof(float), stream);
    edge_att<<<nb(EE * 32), 256, 0, stream>>>(ei, kqvs, bufS);

    // ---- x3 + classifier (writes both tuple copies) ----
    final_fc<<<NN / 8, 160, 0, stream>>>(bufS, kqvs, Wfc, bfc, (float*)d_out);
}
